// TransformerEncoderLayer_15513421873185
// MI455X (gfx1250) — matrix operand order repaired and checked by reading, not compiled, not run on hardware
//
#include <hip/hip_runtime.h>

// ---------------------------------------------------------------------------
// Types for CDNA5 WMMA (wave32): A/B are 16 bf16 per lane, C/D 8 f32 per lane.
// ---------------------------------------------------------------------------
typedef __attribute__((ext_vector_type(16))) __bf16 v16bf;
typedef __attribute__((ext_vector_type(8)))  __bf16 v8bf;
typedef __attribute__((ext_vector_type(8)))  float  v8f;
typedef int v4i __attribute__((vector_size(16)));   // matches async-LDS builtin pointee

#define D_MODEL 1024
#define T_SEQ   2048
#define NBATCH  4
#define NHEADS  16
#define HEAD_DIM 64
#define D_FF    4096

// gfx1250 async global->LDS copy path (ASYNCcnt-tracked), guarded so the
// compile cannot break if this toolchain lacks the builtins.
#if defined(__has_builtin)
#if __has_builtin(__builtin_amdgcn_global_load_async_to_lds_b128) && \
    __has_builtin(__builtin_amdgcn_s_wait_asynccnt)
#define USE_ASYNC_LDS 1
#endif
#endif
#ifndef USE_ASYNC_LDS
#define USE_ASYNC_LDS 0
#endif

__device__ inline v8f zero8() {
    v8f z;
#pragma unroll
    for (int i = 0; i < 8; i++) z[i] = 0.0f;
    return z;
}

// round-to-nearest-even f32 -> bf16 without relying on compiler cvt support
__device__ inline __bf16 f2bf(float f) {
    unsigned u = __builtin_bit_cast(unsigned, f);
    u += 0x7FFFu + ((u >> 16) & 1u);
    unsigned short hs = (unsigned short)(u >> 16);
    return __builtin_bit_cast(__bf16, hs);
}

// build a 16-element bf16 fragment from two 8-element (16B) loads
__device__ inline v16bf ld16(const __bf16* p0, const __bf16* p1) {
    union U { v16bf v; v8bf h[2]; } u;
    u.h[0] = *(const v8bf*)p0;
    u.h[1] = *(const v8bf*)p1;
    return u.v;
}

__device__ inline v8f wmma_bf16(v16bf a, v16bf b, v8f c) {
    // D(16x16 f32) = A(16x32 bf16) * B(32x16 bf16) + C
    return __builtin_amdgcn_wmma_f32_16x16x32_bf16(
        /*neg_a=*/false, a, /*neg_b=*/false, b,
        /*c_mod=*/(short)0, c, /*reuse_a=*/false, /*reuse_b=*/false);
}

#if USE_ASYNC_LDS
// 16-byte async copy global -> LDS (GLOBAL_LOAD_ASYNC_TO_LDS_B128)
__device__ inline void async_cp16(const __bf16* gsrc, __bf16* ldst) {
    __attribute__((address_space(1))) v4i* g =
        (__attribute__((address_space(1))) v4i*)gsrc;
    __attribute__((address_space(3))) v4i* l =
        (__attribute__((address_space(3))) v4i*)ldst;
    __builtin_amdgcn_global_load_async_to_lds_b128(g, l, 0, 0);
}
#endif

// ---------------------------------------------------------------------------
// fp32 -> bf16 weight conversion
// ---------------------------------------------------------------------------
__global__ void f32_to_bf16_kernel(const float* __restrict__ s,
                                   __bf16* __restrict__ d, int n) {
    int base = blockIdx.x * 1024 + threadIdx.x;
#pragma unroll
    for (int j = 0; j < 4; j++) {
        int idx = base + j * 256;
        if (idx < n) d[idx] = f2bf(s[idx]);
    }
}

// ---------------------------------------------------------------------------
// LayerNorm over D=1024, output bf16. One 256-thread block (8 waves) / token.
// ---------------------------------------------------------------------------
__global__ void ln_bf16_kernel(const float* __restrict__ x,
                               const float* __restrict__ g,
                               const float* __restrict__ bta,
                               __bf16* __restrict__ out) {
    __shared__ float red[256];
    const int row = blockIdx.x;
    const int tid = threadIdx.x;
    const float* xr = x + (size_t)row * D_MODEL;

    float v0[4];
    float s = 0.0f;
#pragma unroll
    for (int i = 0; i < 4; i++) { v0[i] = xr[tid + i * 256]; s += v0[i]; }
    red[tid] = s; __syncthreads();
    for (int o = 128; o > 0; o >>= 1) {
        if (tid < o) red[tid] += red[tid + o];
        __syncthreads();
    }
    float mu = red[0] * (1.0f / D_MODEL);
    __syncthreads();

    float s2 = 0.0f;
#pragma unroll
    for (int i = 0; i < 4; i++) { float d = v0[i] - mu; s2 += d * d; }
    red[tid] = s2; __syncthreads();
    for (int o = 128; o > 0; o >>= 1) {
        if (tid < o) red[tid] += red[tid + o];
        __syncthreads();
    }
    float rs = rsqrtf(red[0] * (1.0f / D_MODEL) + 1e-5f);

    __bf16* orow = out + (size_t)row * D_MODEL;
#pragma unroll
    for (int i = 0; i < 4; i++) {
        int c = tid + i * 256;
        orow[c] = f2bf((v0[i] - mu) * rs * g[c] + bta[c]);
    }
}

// ---------------------------------------------------------------------------
// WMMA GEMM: C[M,N] = act( A[M,K]bf16 @ W[N,K]bf16^T + bias[N] (+res) )
// Block tile 128x64, 8 waves in 4x2, each wave 32x32 (2x2 WMMA frags).
// Tile staging uses async global->LDS when available.
// ---------------------------------------------------------------------------
template <bool RELU, bool RES, bool OUTBF>
__global__ void gemm_bf16_kernel(const __bf16* __restrict__ A,
                                 const __bf16* __restrict__ W,
                                 const float*  __restrict__ bias,
                                 const float*  __restrict__ res,
                                 float*        __restrict__ outf,
                                 __bf16*       __restrict__ outb,
                                 int M, int N, int K) {
    __shared__ alignas(32) __bf16 As[128 * 32];
    __shared__ alignas(32) __bf16 Ws[64 * 32];

    const int tid  = threadIdx.x;
    const int lane = tid & 31;
    const int wid  = tid >> 5;
    const int wm   = (wid & 3) * 32;   // wave row offset in block tile
    const int wn   = (wid >> 2) * 32;  // wave col offset in block tile
    const int m0   = blockIdx.y * 128;
    const int n0   = blockIdx.x * 64;

    v8f acc00 = zero8(), acc01 = zero8(), acc10 = zero8(), acc11 = zero8();

    const int srow = tid >> 2;          // 0..63
    const int sseg = (tid & 3) * 8;     // 0,8,16,24
    const int fr   = lane & 15;
    const int hi   = lane >> 4;
    const int akb  = hi * 8;            // A-frag K sub-block
    const int bkb  = hi * 8;            // B-frag K sub-block

    for (int k0 = 0; k0 < K; k0 += 32) {
#if USE_ASYNC_LDS
        // async stage A (128x32): 2 rows/thread, and W (64x32): 1 row/thread
#pragma unroll
        for (int i = 0; i < 2; i++) {
            int row = srow + i * 64;
            async_cp16(&A[(size_t)(m0 + row) * K + k0 + sseg], &As[row * 32 + sseg]);
        }
        async_cp16(&W[(size_t)(n0 + srow) * K + k0 + sseg], &Ws[srow * 32 + sseg]);
        __builtin_amdgcn_s_wait_asynccnt(0);
#else
        // stage A (128x32) : 2 rows per thread
#pragma unroll
        for (int i = 0; i < 2; i++) {
            int row = srow + i * 64;
            *(v8bf*)&As[row * 32 + sseg] =
                *(const v8bf*)&A[(size_t)(m0 + row) * K + k0 + sseg];
        }
        // stage W (64x32) : 1 row per thread
        *(v8bf*)&Ws[srow * 32 + sseg] =
            *(const v8bf*)&W[(size_t)(n0 + srow) * K + k0 + sseg];
#endif
        // prefetch next K tile into caches while this tile computes
        if (k0 + 32 < K) {
            __builtin_prefetch(&A[(size_t)(m0 + srow) * K + k0 + 32 + sseg], 0, 1);
            __builtin_prefetch(&W[(size_t)(n0 + srow) * K + k0 + 32 + sseg], 0, 1);
        }
        __syncthreads();

        const __bf16* abase = &As[wm * 32];
        v16bf a0 = ld16(&abase[fr * 32 + akb],        &abase[fr * 32 + akb + 16]);
        v16bf a1 = ld16(&abase[(fr + 16) * 32 + akb], &abase[(fr + 16) * 32 + akb + 16]);
        const __bf16* bbase = &Ws[wn * 32];
        v16bf b0 = ld16(&bbase[fr * 32 + bkb],        &bbase[fr * 32 + bkb + 16]);
        v16bf b1 = ld16(&bbase[(fr + 16) * 32 + bkb], &bbase[(fr + 16) * 32 + bkb + 16]);

        acc00 = wmma_bf16(a0, b0, acc00);
        acc01 = wmma_bf16(a0, b1, acc01);
        acc10 = wmma_bf16(a1, b0, acc10);
        acc11 = wmma_bf16(a1, b1, acc11);
        __syncthreads();
    }

    // epilogue: C layout -> lane holds N=lane%16, VGPR r holds M = r + 8*(lane/16)
    const int col = lane & 15;
#pragma unroll
    for (int fm = 0; fm < 2; fm++) {
#pragma unroll
        for (int fn = 0; fn < 2; fn++) {
            v8f acc = (fm == 0) ? ((fn == 0) ? acc00 : acc01)
                                : ((fn == 0) ? acc10 : acc11);
            int n  = n0 + wn + fn * 16 + col;
            int mb = m0 + wm + fm * 16 + hi * 8;
            float bv = bias[n];
#pragma unroll
            for (int r = 0; r < 8; r++) {
                size_t idx = (size_t)(mb + r) * N + n;
                float val = acc[r] + bv;
                if (RES)  val += res[idx];
                if (RELU) val = fmaxf(val, 0.0f);
                if (OUTBF) outb[idx] = f2bf(val);
                else       outf[idx] = val;
            }
        }
    }
}

// ---------------------------------------------------------------------------
// Flash attention (bf16 WMMA, online softmax). One wave = 16 queries x Hd=64.
// 4 waves/block share K/V key-range; V staged transposed into LDS.
// ---------------------------------------------------------------------------
__global__ void flash_attn_kernel(const __bf16* __restrict__ q,
                                  const __bf16* __restrict__ k,
                                  const __bf16* __restrict__ v,
                                  __bf16* __restrict__ o) {
    __shared__ alignas(32) __bf16 Vt[64 * 32];      // V^T [hd][key]
    __shared__ alignas(32) __bf16 P[4][16][32];     // per-wave P staging

    const int tid  = threadIdx.x;
    const int lane = tid & 31;
    const int wid  = tid >> 5;
    const int b    = blockIdx.z;
    const int h    = blockIdx.y;
    const int q0   = (blockIdx.x * 4 + wid) * 16;
    const size_t base = ((size_t)b * T_SEQ) * D_MODEL + (size_t)h * HEAD_DIM;
    const int hi  = lane >> 4;
    const int col = lane & 15;

    // Q fragments (A layout): lane = row, K sub-blocks per half-wave
    v16bf qf0, qf1;
    {
        const __bf16* qp = q + base + (size_t)(q0 + col) * D_MODEL;
        int kb = hi * 8;
        qf0 = ld16(qp + kb,      qp + kb + 16);
        qf1 = ld16(qp + 32 + kb, qp + 32 + kb + 16);
    }

    v8f oa[4];
#pragma unroll
    for (int c = 0; c < 4; c++) oa[c] = zero8();
    float m_i[8], l_i[8];
#pragma unroll
    for (int r = 0; r < 8; r++) { m_i[r] = -3.0e38f; l_i[r] = 0.0f; }

    for (int kt = 0; kt < T_SEQ; kt += 32) {
        // --- stage V^T (32 keys x 64 hd) cooperatively ---
#pragma unroll
        for (int i = 0; i < 2; i++) {
            int idx = tid + i * 128;
            int key = idx >> 3;
            int hd0 = (idx & 7) * 8;
            v8bf val = *(const v8bf*)(v + base + (size_t)(kt + key) * D_MODEL + hd0);
#pragma unroll
            for (int j = 0; j < 8; j++) Vt[(hd0 + j) * 32 + key] = val[j];
        }
        __syncthreads();

        // --- S = Q @ K^T (two 16x16 score tiles, contraction over Hd=64) ---
        v8f s0 = zero8(), s1 = zero8();
#pragma unroll
        for (int kk = 0; kk < 2; kk++) {
            const __bf16* kp0 = k + base + (size_t)(kt + col) * D_MODEL + kk * 32 + hi * 8;
            const __bf16* kp1 = k + base + (size_t)(kt + 16 + col) * D_MODEL + kk * 32 + hi * 8;
            v16bf a = (kk == 0) ? qf0 : qf1;
            s0 = wmma_bf16(a, ld16(kp0, kp0 + 16), s0);
            s1 = wmma_bf16(a, ld16(kp1, kp1 + 16), s1);
        }

        // --- online softmax (per-row reductions within 16-lane half-waves) ---
        float al[8];
#pragma unroll
        for (int r = 0; r < 8; r++) {
            float a0 = s0[r] * 0.125f;
            float a1 = s1[r] * 0.125f;
            float rm = fmaxf(a0, a1);
#pragma unroll
            for (int msk = 8; msk >= 1; msk >>= 1)
                rm = fmaxf(rm, __shfl_xor(rm, msk, 32));
            float nm    = fmaxf(m_i[r], rm);
            float alpha = __expf(m_i[r] - nm);
            float e0 = __expf(a0 - nm);
            float e1 = __expf(a1 - nm);
            float rs = e0 + e1;
#pragma unroll
            for (int msk = 8; msk >= 1; msk >>= 1)
                rs += __shfl_xor(rs, msk, 32);
            l_i[r] = l_i[r] * alpha + rs;
            m_i[r] = nm;
            al[r]  = alpha;
            int row = r + 8 * hi;
            P[wid][row][col]      = f2bf(e0);   // keys kt..kt+15
            P[wid][row][col + 16] = f2bf(e1);   // keys kt+16..kt+31
        }
#pragma unroll
        for (int c = 0; c < 4; c++)
#pragma unroll
            for (int r = 0; r < 8; r++) oa[c][r] *= al[r];

        // --- O += P @ V  (A-frag from LDS P, B-frags from Vt) ---
        v16bf pf;
        {
            int kb = hi * 8;
            pf = ld16(&P[wid][col][kb], &P[wid][col][kb + 16]);
        }
#pragma unroll
        for (int c = 0; c < 4; c++) {
            const __bf16* vp = &Vt[(c * 16 + col) * 32 + hi * 8];
            oa[c] = wmma_bf16(pf, ld16(vp, vp + 16), oa[c]);
        }
        __syncthreads();
    }

    // --- normalize and write out (bf16, [B,T,H,Hd] flat == [B*T, D]) ---
#pragma unroll
    for (int c = 0; c < 4; c++) {
#pragma unroll
        for (int r = 0; r < 8; r++) {
            int row = r + 8 * hi;
            int t   = q0 + row;
            o[base + (size_t)t * D_MODEL + c * 16 + col] = f2bf(oa[c][r] / l_i[r]);
        }
    }
}

// ---------------------------------------------------------------------------
// Launcher
// ---------------------------------------------------------------------------
extern "C" void kernel_launch(void* const* d_in, const int* in_sizes, int n_in,
                              void* d_out, int out_size, void* d_ws, size_t ws_size,
                              hipStream_t stream) {
    (void)in_sizes; (void)n_in; (void)out_size; (void)ws_size;

    const float* x   = (const float*)d_in[0];
    const float* Wq  = (const float*)d_in[1];
    const float* bq  = (const float*)d_in[2];
    const float* Wk  = (const float*)d_in[3];
    const float* bk  = (const float*)d_in[4];
    const float* Wv  = (const float*)d_in[5];
    const float* bv  = (const float*)d_in[6];
    const float* Wo  = (const float*)d_in[7];
    const float* bo  = (const float*)d_in[8];
    const float* W1  = (const float*)d_in[9];
    const float* b1  = (const float*)d_in[10];
    const float* W2  = (const float*)d_in[11];
    const float* b2  = (const float*)d_in[12];
    const float* g1  = (const float*)d_in[13];
    const float* be1 = (const float*)d_in[14];
    const float* g2  = (const float*)d_in[15];
    const float* be2 = (const float*)d_in[16];
    float* out = (float*)d_out;

    char* ws = (char*)d_ws;
    size_t off = 0;
    auto alloc = [&](size_t elems) -> __bf16* {
        __bf16* p = (__bf16*)(ws + off);
        off += ((elems * sizeof(__bf16) + 255) & ~(size_t)255);
        return p;
    };

    const size_t DD = (size_t)D_MODEL * D_MODEL;
    const size_t MT = (size_t)NBATCH * T_SEQ;   // 8192 tokens

    __bf16* wq_b  = alloc(DD);
    __bf16* wk_b  = alloc(DD);
    __bf16* wv_b  = alloc(DD);
    __bf16* wo_b  = alloc(DD);
    __bf16* w1_b  = alloc((size_t)D_FF * D_MODEL);
    __bf16* w2_b  = alloc((size_t)D_FF * D_MODEL);
    __bf16* xn_b  = alloc(MT * D_MODEL);
    __bf16* q_b   = alloc(MT * D_MODEL);
    __bf16* k_b   = alloc(MT * D_MODEL);
    __bf16* v_b   = alloc(MT * D_MODEL);
    __bf16* at_b  = alloc(MT * D_MODEL);
    __bf16* xn2_b = alloc(MT * D_MODEL);
    __bf16* h_b   = alloc(MT * D_FF);

    auto conv = [&](const float* s, __bf16* d, size_t n) {
        f32_to_bf16_kernel<<<dim3((unsigned)((n + 1023) / 1024)), dim3(256), 0, stream>>>(
            s, d, (int)n);
    };
    conv(Wq, wq_b, DD);
    conv(Wk, wk_b, DD);
    conv(Wv, wv_b, DD);
    conv(Wo, wo_b, DD);
    conv(W1, w1_b, (size_t)D_FF * D_MODEL);
    conv(W2, w2_b, (size_t)D_FF * D_MODEL);

    // LN1
    ln_bf16_kernel<<<dim3((unsigned)MT), 256, 0, stream>>>(x, g1, be1, xn_b);

    // QKV projections
    dim3 gd(D_MODEL / 64, (unsigned)(MT / 128));
    gemm_bf16_kernel<false, false, true><<<gd, 256, 0, stream>>>(
        xn_b, wq_b, bq, nullptr, nullptr, q_b, (int)MT, D_MODEL, D_MODEL);
    gemm_bf16_kernel<false, false, true><<<gd, 256, 0, stream>>>(
        xn_b, wk_b, bk, nullptr, nullptr, k_b, (int)MT, D_MODEL, D_MODEL);
    gemm_bf16_kernel<false, false, true><<<gd, 256, 0, stream>>>(
        xn_b, wv_b, bv, nullptr, nullptr, v_b, (int)MT, D_MODEL, D_MODEL);

    // flash attention: grid (query tiles / 4 waves, heads, batch)
    flash_attn_kernel<<<dim3(T_SEQ / 64, NHEADS, NBATCH), 128, 0, stream>>>(
        q_b, k_b, v_b, at_b);

    // output projection + residual(x) -> d_out (fp32 residual stream)
    gemm_bf16_kernel<false, true, false><<<gd, 256, 0, stream>>>(
        at_b, wo_b, bo, x, out, nullptr, (int)MT, D_MODEL, D_MODEL);

    // LN2 reads d_out
    ln_bf16_kernel<<<dim3((unsigned)MT), 256, 0, stream>>>(out, g2, be2, xn2_b);

    // FFN1 (ReLU, bf16 out)
    gemm_bf16_kernel<true, false, true><<<dim3(D_FF / 64, (unsigned)(MT / 128)), 256, 0, stream>>>(
        xn2_b, w1_b, b1, nullptr, nullptr, h_b, (int)MT, D_FF, D_MODEL);

    // FFN2 + residual, in-place on d_out (elementwise read-then-write is safe)
    gemm_bf16_kernel<false, true, false><<<dim3(D_MODEL / 64, (unsigned)(MT / 128)), 256, 0, stream>>>(
        h_b, w2_b, b2, out, out, nullptr, (int)MT, D_MODEL, D_FF);
}
